// DelightGPTLanguageModel_38646115729766
// MI455X (gfx1250) — matrix operand order repaired and checked by reading, not compiled, not run on hardware
//
#include <hip/hip_runtime.h>
#include <hip/hip_bf16.h>
#include <math.h>

// ---------------------------------------------------------------------------
// DelightGPT (OpenELM-style GPT) forward on gfx1250 (MI455X).
// bf16 operand residency: weights pre-transposed/converted to bf16 [N][K],
// activations that feed matmuls stored as bf16. Hot loops are pure
// global_load_b128 / ds_load_b128 -> v_wmma_f32_16x16x32_bf16, with the GEMM
// B-tile staged via GLOBAL_LOAD_ASYNC_TO_LDS_B128 (ASYNCcnt).
// ---------------------------------------------------------------------------

typedef __attribute__((ext_vector_type(16))) __bf16 bf16x16;
typedef __attribute__((ext_vector_type(8)))  __bf16 bf16x8;
typedef __attribute__((ext_vector_type(8)))  float  floatx8;

__device__ __forceinline__ bf16x16 concat8(bf16x8 a, bf16x8 b) {
  return __builtin_shufflevector(a, b, 0, 1, 2, 3, 4, 5, 6, 7,
                                 8, 9, 10, 11, 12, 13, 14, 15);
}

// ---------------------------------------------------------------------------
// Weight prep: out[n*K + k] = (bf16) in[k][n], where in is f32 row-major [K,N],
// or head-blocked (h,E,HS) when headStride = E*HS:
//   in index = (n>>6)*headStride + k*64 + (n&63)
// ---------------------------------------------------------------------------
__global__ __launch_bounds__(256)
void transpose_w_k(const float* __restrict__ in, __bf16* __restrict__ out,
                   int K, int N, int headStride) {
  __shared__ float t[16][17];
  int n0 = blockIdx.x * 16, k0 = blockIdx.y * 16;
  int tx = threadIdx.x & 15, ty = threadIdx.x >> 4;
  int n = n0 + tx, k = k0 + ty;
  float v = 0.f;
  if (n < N && k < K)
    v = headStride ? in[(size_t)(n >> 6) * headStride + (size_t)k * 64 + (n & 63)]
                   : in[(size_t)k * N + n];
  t[ty][tx] = v;
  __syncthreads();
  int n2 = n0 + ty, k2 = k0 + tx;
  if (n2 < N && k2 < K)
    out[(size_t)n2 * K + k2] = (__bf16)t[tx][ty];
}

// ---------------------------------------------------------------------------
// Embedding: x[row,:] = tok[idx[row],:] + pos[row % T,:]  (f32 residual stream)
// ---------------------------------------------------------------------------
__global__ void embed_k(const int* __restrict__ idx, const float* __restrict__ tok,
                        const float* __restrict__ pos, float* __restrict__ x,
                        int T_, int E_) {
  int row = blockIdx.x;
  int t = row % T_;
  int id = idx[row];
  for (int c = threadIdx.x; c < E_; c += blockDim.x)
    x[(size_t)row * E_ + c] = tok[(size_t)id * E_ + c] + pos[(size_t)t * E_ + c];
}

// ---------------------------------------------------------------------------
// LayerNorm (biased variance), one block per row; emits bf16 (feeds GEMM A).
// ---------------------------------------------------------------------------
__global__ __launch_bounds__(256)
void layernorm_k(const float* __restrict__ x, const float* __restrict__ g,
                 const float* __restrict__ bta, __bf16* __restrict__ y, int E_) {
  __shared__ float red[256];
  int row = blockIdx.x, tid = threadIdx.x;
  const float* xr = x + (size_t)row * E_;
  float s = 0.f;
  for (int c = tid; c < E_; c += 256) s += xr[c];
  red[tid] = s; __syncthreads();
  for (int o = 128; o; o >>= 1) { if (tid < o) red[tid] += red[tid + o]; __syncthreads(); }
  float mean = red[0] / (float)E_;
  __syncthreads();
  float s2 = 0.f;
  for (int c = tid; c < E_; c += 256) { float d = xr[c] - mean; s2 += d * d; }
  red[tid] = s2; __syncthreads();
  for (int o = 128; o; o >>= 1) { if (tid < o) red[tid] += red[tid + o]; __syncthreads(); }
  float rstd = rsqrtf(red[0] / (float)E_ + 1e-5f);
  for (int c = tid; c < E_; c += 256)
    y[(size_t)row * E_ + c] = (__bf16)((xr[c] - mean) * rstd * g[c] + bta[c]);
}

// ---------------------------------------------------------------------------
// WMMA GEMM: A bf16 [M,K] (lda), Bt bf16 [N][K] (pre-transposed, ld = K).
// Tile 128x64x32, 256 threads (8 waves).
//  - A-frag: two direct global 16B loads per lane (rows are wave-exclusive).
//  - B tile: one GLOBAL_LOAD_ASYNC_TO_LDS_B128 per thread per K-step
//    (no VGPR round-trip; tracked with ASYNCcnt).
// Epilogue: + bias(f32) + res(f32), optional relu; writes f32 C and/or bf16 C;
// optional transposed-V scatter (vout[b][h][d][t], T=1024) for attention.
// Requires M % 128 == 0 (true here: M = 4096).
// ---------------------------------------------------------------------------
#define BM 128
#define BN 64
#define BKK 32

__global__ __launch_bounds__(256)
void gemm_wmma(const __bf16* __restrict__ A, const __bf16* __restrict__ Bt,
               const float* __restrict__ bias, const float* __restrict__ res,
               float* __restrict__ Cf, __bf16* __restrict__ Cb,
               __bf16* __restrict__ vout, int vH,
               int M, int N, int K, int lda, int ldc, int relu) {
  __shared__ alignas(16) __bf16 BsT[BN][40];  // [n][k], stride 80B
  const int tid  = threadIdx.x;
  const int lane = tid & 31;
  const int wave = tid >> 5;
  const int ln16 = lane & 15;
  const int hi   = (lane >> 4) & 1;
  const int m0 = blockIdx.y * BM, n0 = blockIdx.x * BN;

  // B stage assignment: 64 rows x 4 16B-chunks = 256 -> one async b128 each.
  const int snl = tid >> 2, sch = (tid & 3) * 8;
  int sgn = n0 + snl; if (sgn >= N) sgn = N - 1;  // clamp; ragged N guarded at store
  const __bf16* srow = Bt + (size_t)sgn * K + sch;
  // LDS byte address (addrspace(3)) of this thread's staging slot.
  const unsigned ldsDst =
      (unsigned)(size_t)(__attribute__((address_space(3))) void*)&BsT[snl][sch];

  const __bf16* arow = A + (size_t)(m0 + wave * 16 + ln16) * lda;

  floatx8 acc[4] = {};

  for (int k0 = 0; k0 < K; k0 += BKK) {
    // Async copy B tile chunk: memory -> LDS, no VGPR round-trip.
    asm volatile("global_load_async_to_lds_b128 %0, %1, off"
                 :: "v"(ldsDst), "v"(srow + k0) : "memory");
    if (k0 + BKK < K)  // prefetch next A K-slice (global_prefetch_b8)
      __builtin_prefetch(arow + ((tid & 7) << 4) * lda + k0 + BKK, 0, 3);
    asm volatile("s_wait_asynccnt 0x0" ::: "memory");
    __syncthreads();

    bf16x16 af = concat8(*(const bf16x8*)(arow + k0 + 16 * hi),
                         *(const bf16x8*)(arow + k0 + 16 * hi + 8));
#pragma unroll
    for (int nt = 0; nt < 4; ++nt) {
      const __bf16* bp = &BsT[nt * 16 + ln16][16 * hi];
      bf16x16 bfrag = concat8(*(const bf16x8*)bp, *(const bf16x8*)(bp + 8));
      acc[nt] = __builtin_amdgcn_wmma_f32_16x16x32_bf16(
          false, af, false, bfrag, (short)0, acc[nt], false, false);
    }
    __syncthreads();
  }

  // Epilogue: C-frag row = wave*16 + r + 8*hi, col = nt*16 + lane%16.
#pragma unroll
  for (int nt = 0; nt < 4; ++nt) {
#pragma unroll
    for (int r = 0; r < 8; ++r) {
      int gm = m0 + wave * 16 + r + 8 * hi;
      int gn = n0 + nt * 16 + ln16;
      if (gn < N) {
        float vv = acc[nt][r];
        if (bias) vv += bias[gn];
        if (res)  vv += res[(size_t)gm * ldc + gn];
        if (relu) vv = fmaxf(vv, 0.f);
        if (Cf) Cf[(size_t)gm * ldc + gn] = vv;
        if (Cb) Cb[(size_t)gm * N + gn] = (__bf16)vv;
        if (vout) {  // scatter transposed V: [b][h][d][t], T = 1024
          int bb = gm >> 10, t = gm & 1023, hh = gn >> 6, d = gn & 63;
          vout[((((size_t)bb * vH + hh) << 6) + d) * 1024 + t] = (__bf16)vv;
        }
      }
    }
  }
}

// ---------------------------------------------------------------------------
// Flash attention, causal, one wave per (b, h, 16-query tile).
// q/k bf16 [b*T+t][h*64+d] (ld = H*64); vt bf16 [b][h][d][t]; out ao bf16.
// All WMMA operands are direct global b128 loads except the tiny P buffer.
// Requires T % 32 == 0 (key loads never OOB).
// ---------------------------------------------------------------------------
__global__ __launch_bounds__(32)
void attn_flash(const __bf16* __restrict__ q, const __bf16* __restrict__ k,
                const __bf16* __restrict__ vt, __bf16* __restrict__ o,
                int T_, int H, int ld) {
  __shared__ alignas(16) __bf16 Pb[16][32];  // [q_local][key_local] (A-frag)
  const int lane = threadIdx.x & 31;
  const int ln16 = lane & 15;
  const int hi   = (lane >> 4) & 1;
  const int b = blockIdx.z, h = blockIdx.y, q0 = blockIdx.x * 16;
  const size_t bT = (size_t)b * T_;

  // Q fragments: two 32-wide d-chunks, direct bf16 loads.
  bf16x16 aq[2];
  {
    const __bf16* qr = q + (bT + q0 + ln16) * ld + h * 64;
#pragma unroll
    for (int kc = 0; kc < 2; ++kc) {
      const __bf16* c0 = qr + kc * 32 + 8 * hi;
      aq[kc] = concat8(*(const bf16x8*)c0, *(const bf16x8*)(c0 + 16));
    }
  }

  float m_[8], l_[8];
#pragma unroll
  for (int r = 0; r < 8; ++r) { m_[r] = -1e30f; l_[r] = 0.f; }
  floatx8 oacc[4] = {};

  const size_t vbase = (((size_t)b * H + h) << 6) * 1024;  // vt[b][h][0][0]

  for (int s0 = 0; s0 <= q0 + 15; s0 += 32) {
    // Scores S[16 x 32 keys]: K^T B-frags loaded direct from global bf16.
    floatx8 sc[2] = {};
#pragma unroll
    for (int n2 = 0; n2 < 2; ++n2) {
      const __bf16* krow = k + (bT + s0 + n2 * 16 + ln16) * ld + h * 64;
#pragma unroll
      for (int kc = 0; kc < 2; ++kc) {
        const __bf16* kp = krow + kc * 32 + 8 * hi;
        bf16x16 bk = concat8(*(const bf16x8*)kp, *(const bf16x8*)(kp + 16));
        sc[n2] = __builtin_amdgcn_wmma_f32_16x16x32_bf16(
            false, aq[kc], false, bk, (short)0, sc[n2], false, false);
      }
    }

    // Scale + causal mask + online softmax; P -> LDS bf16 (A-frag layout).
#pragma unroll
    for (int r = 0; r < 8; ++r) {
      int qi = q0 + r + 8 * hi;
      float s0v = sc[0][r] * 0.125f, s1v = sc[1][r] * 0.125f;
      if (s0 + ln16 > qi)      s0v = -1e30f;
      if (s0 + 16 + ln16 > qi) s1v = -1e30f;
      float tmax = fmaxf(s0v, s1v);
      for (int off = 8; off; off >>= 1)
        tmax = fmaxf(tmax, __shfl_xor(tmax, off, 32));
      float mn = fmaxf(m_[r], tmax);
      float fr = __expf(m_[r] - mn);
      float p0 = __expf(s0v - mn), p1 = __expf(s1v - mn);
      float ps = p0 + p1;
      for (int off = 8; off; off >>= 1)
        ps += __shfl_xor(ps, off, 32);
      l_[r] = l_[r] * fr + ps;
      m_[r] = mn;
#pragma unroll
      for (int f = 0; f < 4; ++f) oacc[f][r] = oacc[f][r] * fr;
      int prow = r + 8 * hi;
      Pb[prow][ln16]      = (__bf16)p0;
      Pb[prow][16 + ln16] = (__bf16)p1;
    }
    __syncthreads();

    // O += P(16x32) * V(32x64); V B-frags direct from transposed vt.
    bf16x16 ap = concat8(*(const bf16x8*)&Pb[ln16][8 * hi],
                         *(const bf16x8*)&Pb[ln16][16 + 8 * hi]);
#pragma unroll
    for (int n4 = 0; n4 < 4; ++n4) {
      const __bf16* vrow = vt + vbase + (size_t)(n4 * 16 + ln16) * 1024 + s0 + 8 * hi;
      bf16x16 bv = concat8(*(const bf16x8*)vrow, *(const bf16x8*)(vrow + 16));
      oacc[n4] = __builtin_amdgcn_wmma_f32_16x16x32_bf16(
          false, ap, false, bv, (short)0, oacc[n4], false, false);
    }
    __syncthreads();  // Pb reused next iteration
  }

#pragma unroll
  for (int r = 0; r < 8; ++r) {
    int row = q0 + r + 8 * hi;
    float inv = 1.f / l_[r];
#pragma unroll
    for (int n4 = 0; n4 < 4; ++n4)
      o[(bT + row) * ld + h * 64 + n4 * 16 + ln16] = (__bf16)(oacc[n4][r] * inv);
  }
}

// ---------------------------------------------------------------------------
// Loss: mean over rows of (logsumexp(logits[row]) - logits[row, target]).
// ---------------------------------------------------------------------------
__global__ void zero_loss_k(float* loss) {
  if (threadIdx.x == 0 && blockIdx.x == 0) *loss = 0.f;
}

__global__ __launch_bounds__(256)
void loss_k(const float* __restrict__ logits, const int* __restrict__ tgt,
            float* __restrict__ loss, int V_, int NR) {
  __shared__ float red[256];
  int row = blockIdx.x, tid = threadIdx.x;
  const float* lr = logits + (size_t)row * V_;
  float mx = -1e30f;
  for (int c = tid; c < V_; c += 256) mx = fmaxf(mx, lr[c]);
  red[tid] = mx; __syncthreads();
  for (int o = 128; o; o >>= 1) { if (tid < o) red[tid] = fmaxf(red[tid], red[tid + o]); __syncthreads(); }
  mx = red[0]; __syncthreads();
  float se = 0.f;
  for (int c = tid; c < V_; c += 256) se += __expf(lr[c] - mx);
  red[tid] = se; __syncthreads();
  for (int o = 128; o; o >>= 1) { if (tid < o) red[tid] += red[tid + o]; __syncthreads(); }
  if (tid == 0) {
    float lse = logf(red[0]) + mx;
    atomicAdd(loss, (lse - lr[tgt[row]]) / (float)NR);
  }
}

// ---------------------------------------------------------------------------
// Host orchestration.
// ---------------------------------------------------------------------------
extern "C" void kernel_launch(void* const* d_in, const int* in_sizes, int n_in,
                              void* d_out, int out_size, void* d_ws, size_t ws_size,
                              hipStream_t stream) {
  (void)in_sizes; (void)n_in; (void)out_size; (void)ws_size;

  const int V = 50257, E = 768, HS = 64, T = 1024, Bb = 4, NL = 8;
  const int M = Bb * T;  // 4096 rows
  static const int heads[NL] = {6, 7, 8, 9, 9, 10, 11, 12};
  static const int ffn[NL]   = {512, 768, 1280, 1536, 2048, 2304, 2816, 3072};
  const int FMAX = 3072;

  const int*   idx = (const int*)d_in[0];
  const int*   tgt = (const int*)d_in[1];
  const float* tok = (const float*)d_in[2];
  const float* pos = (const float*)d_in[3];
  auto blk = [&](int l, int j) { return (const float*)d_in[4 + l * 13 + j]; };
  // j: 0 wq, 1 wk, 2 wv, 3 proj_w, 4 proj_b, 5 ln1_g, 6 ln1_b, 7 ln2_g,
  //    8 ln2_b, 9 w1, 10 b1, 11 w2, 12 b2
  const float* lnf_g = (const float*)d_in[4 + NL * 13 + 0];
  const float* lnf_b = (const float*)d_in[4 + NL * 13 + 1];
  const float* lm_w  = (const float*)d_in[4 + NL * 13 + 2];
  const float* lm_b  = (const float*)d_in[4 + NL * 13 + 3];

  // Workspace layout.
  const size_t SZF  = (size_t)M * E * sizeof(float);    // 12.6 MB f32
  const size_t SZB  = (size_t)M * E * sizeof(__bf16);   // 6.3 MB bf16
  char* w = (char*)d_ws;
  float*  x    = (float*)w;              w += SZF;   // residual stream
  float*  xn   = (float*)w;              w += SZF;   // post-attention residual
  __bf16* hln  = (__bf16*)w;             w += SZB;   // LN output (GEMM A)
  __bf16* qb   = (__bf16*)w;             w += SZB;
  __bf16* kb   = (__bf16*)w;             w += SZB;
  __bf16* vt   = (__bf16*)w;             w += (size_t)Bb * 12 * 64 * T * 2;  // [b][h][d][t]
  __bf16* ao   = (__bf16*)w;             w += SZB;
  __bf16* ff   = (__bf16*)w;             w += (size_t)M * FMAX * 2;
  __bf16* wqT  = (__bf16*)w;             w += (size_t)E * E * 2;   // [KN][E] max
  __bf16* wkT  = (__bf16*)w;             w += (size_t)E * E * 2;
  __bf16* wvT  = (__bf16*)w;             w += (size_t)E * E * 2;
  __bf16* prT  = (__bf16*)w;             w += (size_t)E * E * 2;   // [E][KN]
  __bf16* w1T  = (__bf16*)w;             w += (size_t)FMAX * E * 2;
  __bf16* w2T  = (__bf16*)w;             w += (size_t)E * FMAX * 2;
  __bf16* lmT  = (__bf16*)w;             w += (size_t)V * E * 2;   // 77 MB

  auto transp = [&](const float* in, __bf16* out, int K_, int N_, int hs) {
    dim3 g((N_ + 15) / 16, (K_ + 15) / 16);
    transpose_w_k<<<g, dim3(256), 0, stream>>>(in, out, K_, N_, hs);
  };
  auto gemm = [&](const __bf16* A, const __bf16* Bt, const float* bias,
                  const float* res, float* Cf, __bf16* Cb, __bf16* vo, int vH,
                  int M_, int N_, int K_, int lda, int ldc, int relu) {
    dim3 g((N_ + BN - 1) / BN, (M_ + BM - 1) / BM);
    gemm_wmma<<<g, dim3(256), 0, stream>>>(A, Bt, bias, res, Cf, Cb, vo, vH,
                                           M_, N_, K_, lda, ldc, relu);
  };

  embed_k<<<dim3(M), dim3(256), 0, stream>>>(idx, tok, pos, x, T, E);

  for (int l = 0; l < NL; ++l) {
    const int H = heads[l], F = ffn[l], KN = H * HS;
    // Weight prep (bf16 [N][K]).
    transp(blk(l, 0), wqT, E, KN, E * HS);
    transp(blk(l, 1), wkT, E, KN, E * HS);
    transp(blk(l, 2), wvT, E, KN, E * HS);
    transp(blk(l, 3), prT, KN, E, 0);
    transp(blk(l, 9), w1T, E, F, 0);
    transp(blk(l, 11), w2T, F, E, 0);

    layernorm_k<<<dim3(M), dim3(256), 0, stream>>>(x, blk(l, 5), blk(l, 6), hln, E);
    gemm(hln, wqT, nullptr, nullptr, nullptr, qb, nullptr, 0, M, KN, E, E, 0, 0);
    gemm(hln, wkT, nullptr, nullptr, nullptr, kb, nullptr, 0, M, KN, E, E, 0, 0);
    gemm(hln, wvT, nullptr, nullptr, nullptr, nullptr, vt, H, M, KN, E, E, 0, 0);
    attn_flash<<<dim3(T / 16, H, Bb), dim3(32), 0, stream>>>(qb, kb, vt, ao, T, H, KN);
    gemm(ao, prT, blk(l, 4), x, xn, nullptr, nullptr, 0, M, E, KN, KN, E, 0);
    layernorm_k<<<dim3(M), dim3(256), 0, stream>>>(xn, blk(l, 7), blk(l, 8), hln, E);
    gemm(hln, w1T, blk(l, 10), nullptr, nullptr, ff, nullptr, 0, M, F, E, E, 0, 1);
    gemm(ff, w2T, blk(l, 12), xn, x, nullptr, nullptr, 0, M, E, F, F, E, 0);
  }

  layernorm_k<<<dim3(M), dim3(256), 0, stream>>>(x, lnf_g, lnf_b, hln, E);
  transp(lm_w, lmT, E, V, 0);
  float* logits = (float*)d_out;
  gemm(hln, lmT, lm_b, nullptr, logits, nullptr, nullptr, 0, M, V, E, E, V, 0);

  float* loss = logits + (size_t)M * V;
  zero_loss_k<<<dim3(1), dim3(32), 0, stream>>>(loss);
  loss_k<<<dim3(M), dim3(256), 0, stream>>>(logits, tgt, loss, V, M);
}
